// Trans_36515811950804
// MI455X (gfx1250) — compile-verified, matrix operand order checked by reading, not run on hardware
//
#include <hip/hip_runtime.h>
#include <hip/hip_bf16.h>

// GNN with 3 TransformerConv layers (heads=1), global mean pool, linear head.
// N=100000, E=1600000, B=64, F_IN=128, H=64, C=10.
// Dense projections: f16 WMMA (v_wmma_f32_16x16x32_f16), f32 accumulate.
// A tile staged in LDS (padded, conflict-free) and shared by all 4 waves;
// each wave owns one weight matrix (q/k/v/s) and 2 row-tiles (32 rows/block).

typedef __attribute__((ext_vector_type(16))) _Float16 v16h;
typedef __attribute__((ext_vector_type(8)))  _Float16 v8h;
typedef __attribute__((ext_vector_type(8)))  float    v8f;

#define FIN 128
#define HH  64
#define BB  64
#define CC  10
#define WSLOT (64*128)   // halves per f16 weight slot (max 64x128)
#define RT   2           // row tiles (of 16) per wave
#define PAD  8           // LDS row padding in halves (16B) -> conflict-free b128 reads

// ---------------------------------------------------------------- conversions
__global__ void cvt_f16_kernel(const float* __restrict__ in, _Float16* __restrict__ out, int n) {
    int i = blockIdx.x * blockDim.x + threadIdx.x;
    if (i < n) out[i] = (_Float16)in[i];
}

// ---------------------------------------------------------------- WMMA GEMM
// Block = 128 threads = 4 waves; block covers rows [row0, row0+32).
// Wave w computes Out_w = A_f16[32 x K] @ W_w[64 x K]^T + b_w for w in {q,k,v,s}.
struct QKVS {
    const _Float16* A;
    const _Float16* W0; const _Float16* W1; const _Float16* W2; const _Float16* W3;
    const float* b0; const float* b1; const float* b2; const float* b3;
    float* o0; float* o1; float* o2; float* o3;
    int K;
};

__global__ __launch_bounds__(128)
void qkvs_wmma_kernel(QKVS a) {
    __shared__ __align__(16) _Float16 As[RT * 16 * (FIN + PAD)];  // 32*136 halves = 8704B max
    const int K      = a.K;
    const int stride = K + PAD;
    const int row0   = blockIdx.x * (RT * 16);
    const int tid    = threadIdx.x;

    // ---- cooperative stage of A rows into LDS (16B chunks), padded row stride
    const int chunks_per_row = K >> 3;
    for (int i = tid; i < RT * 16 * chunks_per_row; i += 128) {
        int r  = i / chunks_per_row;
        int ch = i - r * chunks_per_row;
        *(v8h*)(As + r * stride + ch * 8) =
            *(const v8h*)(a.A + (size_t)(row0 + r) * K + ch * 8);
    }
    __syncthreads();

    const int wave = tid >> 5;
    const int lane = tid & 31;
    const int half = lane >> 4;        // 0: K 0-7/16-23, 1: K 8-15/24-31 (ISA 7.12.2)
    const int l16  = lane & 15;

    const _Float16* W  = (wave==0) ? a.W0 : (wave==1) ? a.W1 : (wave==2) ? a.W2 : a.W3;
    const float*    bs = (wave==0) ? a.b0 : (wave==1) ? a.b1 : (wave==2) ? a.b2 : a.b3;
    float*          O  = (wave==0) ? a.o0 : (wave==1) ? a.o1 : (wave==2) ? a.o2 : a.o3;

    v8f acc[RT][4] = {};               // RT row tiles x 4 col tiles (16x16 f32 each)

    for (int k0 = 0; k0 < K; k0 += 32) {
        const int kb = k0 + half * 8;

        // B fragments for all 4 col tiles, reused across row tiles.
        // B = W^T: column n of B is row n of row-major W -> contiguous 16B loads.
        v16h bf[4];
        #pragma unroll
        for (int nt = 0; nt < 4; ++nt) {
            const _Float16* wrow = W + (size_t)(nt*16 + l16) * K;
            v8h blo = *(const v8h*)(wrow + kb);
            v8h bhi = *(const v8h*)(wrow + kb + 16);
            #pragma unroll
            for (int i = 0; i < 8; ++i) { bf[nt][i] = blo[i]; bf[nt][8+i] = bhi[i]; }
        }

        #pragma unroll
        for (int rt = 0; rt < RT; ++rt) {
            const _Float16* arow = As + (rt*16 + l16) * stride;   // LDS, conflict-free
            v8h alo = *(const v8h*)(arow + kb);
            v8h ahi = *(const v8h*)(arow + kb + 16);
            v16h af;
            #pragma unroll
            for (int i = 0; i < 8; ++i) { af[i] = alo[i]; af[8+i] = ahi[i]; }

            #pragma unroll
            for (int nt = 0; nt < 4; ++nt)
                acc[rt][nt] = __builtin_amdgcn_wmma_f32_16x16x32_f16(
                    false, af, false, bf[nt], (short)0, acc[rt][nt], false, false);
        }
    }

    // C/D layout: lane -> col = l16, rows = half*8 + r
    const int rbase = half * 8;
    #pragma unroll
    for (int rt = 0; rt < RT; ++rt) {
        #pragma unroll
        for (int nt = 0; nt < 4; ++nt) {
            const int col = nt*16 + l16;
            const float bv = bs[col];
            #pragma unroll
            for (int r = 0; r < 8; ++r)
                O[(size_t)(row0 + rt*16 + rbase + r) * HH + col] = acc[rt][nt][r] + bv;
        }
    }
}

// ---------------------------------------------------------------- edge phase
__device__ __forceinline__ unsigned f2ord(float f) {
    unsigned u = __float_as_uint(f);
    return (u & 0x80000000u) ? ~u : (u | 0x80000000u);
}
__device__ __forceinline__ float ord2f(unsigned k) {
    return (k & 0x80000000u) ? __uint_as_float(k ^ 0x80000000u) : __uint_as_float(~k);
}

__global__ void edge_alpha_max_kernel(const float* __restrict__ q, const float* __restrict__ kk,
                                      const int* __restrict__ src, const int* __restrict__ dst,
                                      float* __restrict__ alpha, unsigned* __restrict__ mbits, int E) {
    int e = blockIdx.x * blockDim.x + threadIdx.x;
    if (e >= E) return;
    int s = src[e], d = dst[e];
    const float4* qr = (const float4*)(q  + (size_t)d * HH);
    const float4* kr = (const float4*)(kk + (size_t)s * HH);
    float acc = 0.f;
    #pragma unroll
    for (int i = 0; i < HH/4; ++i) {
        float4 a = qr[i], b = kr[i];
        acc += a.x*b.x + a.y*b.y + a.z*b.z + a.w*b.w;
    }
    acc *= 0.125f;                    // 1/sqrt(64)
    alpha[e] = acc;
    atomicMax(mbits + d, f2ord(acc));
}

__global__ void edge_exp_sum_kernel(float* __restrict__ alpha_eexp, const int* __restrict__ dst,
                                    const unsigned* __restrict__ mbits, float* __restrict__ ssum, int E) {
    int e = blockIdx.x * blockDim.x + threadIdx.x;
    if (e >= E) return;
    int d = dst[e];
    float ex = __expf(alpha_eexp[e] - ord2f(mbits[d]));
    alpha_eexp[e] = ex;               // in-place: alpha buffer becomes e[edge]
    atomicAdd(ssum + d, ex);
}

// 8 lanes per edge, 8 floats each, to spread the atomic traffic.
__global__ void edge_aggregate_kernel(const float* __restrict__ eexp, const float* __restrict__ ssum,
                                      const float* __restrict__ v, const int* __restrict__ src,
                                      const int* __restrict__ dst, float* __restrict__ agg, int E) {
    long long gid = (long long)blockIdx.x * blockDim.x + threadIdx.x;
    if (gid >= (long long)E * 8) return;
    int e = (int)(gid >> 3);
    int c = (int)(gid & 7);
    int s = src[e], d = dst[e];
    float coef = eexp[e] / ssum[d];
    const float* vr = v   + (size_t)s * HH + c * 8;
    float*       ar = agg + (size_t)d * HH + c * 8;
    #pragma unroll
    for (int i = 0; i < 8; ++i) atomicAdd(ar + i, coef * vr[i]);
}

__global__ void add_skip_kernel(const float* __restrict__ agg, const float* __restrict__ slin,
                                float* __restrict__ hout, _Float16* __restrict__ hout16,
                                int n, int relu) {
    int i = blockIdx.x * blockDim.x + threadIdx.x;
    if (i >= n) return;
    float v = agg[i] + slin[i];
    if (relu) v = fmaxf(v, 0.f);
    hout[i]   = v;
    hout16[i] = (_Float16)v;          // f16 copy feeds next layer's WMMA GEMM
}

// ---------------------------------------------------------------- pooling + head
__global__ void pool_kernel(const float* __restrict__ h, const int* __restrict__ batch,
                            float* __restrict__ sums, float* __restrict__ cnts, int N) {
    long long gid = (long long)blockIdx.x * blockDim.x + threadIdx.x;
    if (gid >= (long long)N * 8) return;
    int n = (int)(gid >> 3);
    int c = (int)(gid & 7);
    int b = batch[n];
    if (c == 0) atomicAdd(cnts + b, 1.0f);
    const float* hr = h    + (size_t)n * HH + c * 8;
    float*       sr = sums + (size_t)b * HH + c * 8;
    #pragma unroll
    for (int i = 0; i < 8; ++i) atomicAdd(sr + i, hr[i]);
}

__global__ void final_kernel(const float* __restrict__ sums, const float* __restrict__ cnts,
                             const float* __restrict__ Wl, const float* __restrict__ bl,
                             float* __restrict__ out) {
    int idx = blockIdx.x * blockDim.x + threadIdx.x;
    if (idx >= BB * CC) return;
    int b = idx / CC, c = idx % CC;
    float inv = 1.0f / fmaxf(cnts[b], 1.0f);
    const float* sr = sums + (size_t)b * HH;
    const float* wr = Wl   + (size_t)c * HH;
    float acc = bl[c];
    #pragma unroll
    for (int i = 0; i < HH; ++i) acc += (sr[i] * inv) * wr[i];
    out[idx] = acc;
}

// ---------------------------------------------------------------- launch
extern "C" void kernel_launch(void* const* d_in, const int* in_sizes, int n_in,
                              void* d_out, int out_size, void* d_ws, size_t ws_size,
                              hipStream_t stream) {
    const float* x     = (const float*)d_in[0];
    const int*   ei    = (const int*)d_in[1];
    const int*   batch = (const int*)d_in[2];
    const int N = in_sizes[0] / FIN;          // 100000 (multiple of 32)
    const int E = in_sizes[1] / 2;            // 1600000
    const int* src = ei;
    const int* dst = ei + E;

    const float* W[3][4]; const float* bias_[3][4];
    int idx = 3;
    for (int l = 0; l < 3; ++l)
        for (int j = 0; j < 4; ++j) {         // order: q,k,v,s -> W then b
            W[l][j]     = (const float*)d_in[idx++];
            bias_[l][j] = (const float*)d_in[idx++];
        }
    const float* Wl = (const float*)d_in[27];
    const float* bl = (const float*)d_in[28];

    // ---- carve workspace (256B aligned)
    char* p = (char*)d_ws;
    auto carve = [&](size_t bytes) -> char* {
        char* r = p; p += (bytes + 255) & ~(size_t)255; return r;
    };
    _Float16* xh    = (_Float16*)carve((size_t)N * FIN * sizeof(_Float16)); // f16 activations
    _Float16* Wh    = (_Float16*)carve((size_t)4 * WSLOT * sizeof(_Float16));
    float*    q     = (float*)carve((size_t)N * HH * 4);   // also reused as f32 h-out
    float*    k     = (float*)carve((size_t)N * HH * 4);
    float*    v     = (float*)carve((size_t)N * HH * 4);
    float*    slin  = (float*)carve((size_t)N * HH * 4);
    float*    agg   = (float*)carve((size_t)N * HH * 4);
    float*    alpha = (float*)carve((size_t)E * 4);        // reused in-place as e[edge]
    unsigned* mbits = (unsigned*)carve((size_t)N * 4);
    float*    ssum  = (float*)carve((size_t)N * 4);
    float*    sums  = (float*)carve((size_t)BB * HH * 4);
    float*    cnts  = (float*)carve((size_t)BB * 4);
    (void)ws_size; (void)n_in; (void)out_size;

    const int TB = 256;

    { // x -> f16 once
        int n = N * FIN;
        cvt_f16_kernel<<<(n + TB - 1) / TB, TB, 0, stream>>>(x, xh, n);
    }

    for (int l = 0; l < 3; ++l) {
        const int K = (l == 0) ? FIN : HH;

        for (int j = 0; j < 4; ++j) {         // tiny weight conversions
            int n = HH * K;
            cvt_f16_kernel<<<(n + TB - 1) / TB, TB, 0, stream>>>(W[l][j], Wh + j * WSLOT, n);
        }
        hipMemsetAsync(mbits, 0, (size_t)N * 4, stream);
        hipMemsetAsync(ssum,  0, (size_t)N * 4, stream);
        hipMemsetAsync(agg,   0, (size_t)N * HH * 4, stream);

        QKVS a;
        a.A  = xh; a.K = K;
        a.W0 = Wh;           a.W1 = Wh + WSLOT;   a.W2 = Wh + 2*WSLOT; a.W3 = Wh + 3*WSLOT;
        a.b0 = bias_[l][0];  a.b1 = bias_[l][1];  a.b2 = bias_[l][2];  a.b3 = bias_[l][3];
        a.o0 = q; a.o1 = k; a.o2 = v; a.o3 = slin;
        qkvs_wmma_kernel<<<N / (RT * 16), 128, 0, stream>>>(a);

        edge_alpha_max_kernel<<<(E + TB - 1) / TB, TB, 0, stream>>>(q, k, src, dst, alpha, mbits, E);
        edge_exp_sum_kernel  <<<(E + TB - 1) / TB, TB, 0, stream>>>(alpha, dst, mbits, ssum, E);
        long long tot = (long long)E * 8;
        edge_aggregate_kernel<<<(int)((tot + TB - 1) / TB), TB, 0, stream>>>(alpha, ssum, v, src, dst, agg, E);

        int n = N * HH;   // h_out(f32) -> q buffer (dead here); f16 copy -> xh for next GEMM
        add_skip_kernel<<<(n + TB - 1) / TB, TB, 0, stream>>>(agg, slin, q, xh, n, (l < 2) ? 1 : 0);
    }

    hipMemsetAsync(sums, 0, (size_t)BB * HH * 4, stream);
    hipMemsetAsync(cnts, 0, (size_t)BB * 4, stream);
    long long tot = (long long)N * 8;
    pool_kernel<<<(int)((tot + TB - 1) / TB), TB, 0, stream>>>(q, batch, sums, cnts, N);
    final_kernel<<<(BB * CC + 127) / 128, 128, 0, stream>>>(sums, cnts, Wl, bl, (float*)d_out);
}
